// SiameseSNN_13159779795203
// MI455X (gfx1250) — compile-verified, matrix operand order checked by reading, not run on hardware
//
#include <hip/hip_runtime.h>
#include <hip/hip_fp16.h>

typedef __attribute__((ext_vector_type(16))) _Float16 v16h;
typedef __attribute__((ext_vector_type(8)))  _Float16 v8h;
typedef __attribute__((ext_vector_type(8)))  float    v8f;

#define BETA  0.9f
#define VTH   1.2f
#define BNEPS 1e-5f

// Sizes
#define NB   128      // batch
#define C1   16
#define L1   1981
#define L1P  1984     // padded stride (mult of 4) for float4 access
#define LP   495      // pooled length
#define C2   32
#define L2L  493      // conv2 out length
#define NF   15776    // 32*493 flatten
#define NH   256
#define NE   128
#define KIT  493      // NF/32 k-iterations
#define STEPS 50

// ---------------- fp32 -> fp16 weight conversion (once) ----------------
__global__ void k_w2half(const float* __restrict__ w, _Float16* __restrict__ wh, int n) {
    int i = blockIdx.x * 256 + threadIdx.x;
    if (i < n) wh[i] = (_Float16)w[i];
}

// ---------------- conv1 (stride 4, K=80) + bn1 -> a1 (once/input) ------
__global__ void k_conv1(const float* __restrict__ x, const float* __restrict__ w1,
                        const float* __restrict__ cb,
                        const float* __restrict__ g, const float* __restrict__ bb,
                        const float* __restrict__ rm, const float* __restrict__ rv,
                        float* __restrict__ a1) {
    __shared__ float sw[C1 * 80];
    for (int i = threadIdx.x; i < C1 * 80; i += 256) sw[i] = w1[i];
    __syncthreads();
    int idx = blockIdx.x * 256 + threadIdx.x;
    if (idx >= NB * C1 * L1) return;
    int l = idx % L1; int t = idx / L1;
    int oc = t & 15;  int b = t >> 4;
    const float* xp = x + b * 8000 + l * 4;
    const float* wp = sw + oc * 80;
    float s = 0.f;
#pragma unroll 8
    for (int k = 0; k < 80; ++k) s = fmaf(xp[k], wp[k], s);
    float inv = g[oc] * rsqrtf(rv[oc] + BNEPS);
    a1[(size_t)t * L1P + l] = fmaf(3.0f * s + cb[oc], inv, bb[oc] - rm[oc] * inv);
}

// ---------------- per step: LIF1 + maxpool4 -> p -----------------------
__global__ void k_lif1_pool(const float* __restrict__ a1, float* __restrict__ m1,
                            float* __restrict__ p) {
    int idx = blockIdx.x * 256 + threadIdx.x;
    if (idx >= NB * C1 * LP) return;
    int pp = idx % LP; int t = idx / LP;           // t = b*16 + c
    const float4* av = (const float4*)(a1 + (size_t)t * L1P + pp * 4);
    float4*       mv = (float4*)(m1 + (size_t)t * L1P + pp * 4);
    float4 a = *av; float4 m = *mv;
    float aa[4] = {a.x, a.y, a.z, a.w};
    float mm[4] = {m.x, m.y, m.z, m.w};
    float mx = 0.f;
#pragma unroll
    for (int k = 0; k < 4; ++k) {
        float reset = (mm[k] > VTH) ? 1.f : 0.f;
        float nm = fmaf(BETA, mm[k], aa[k]) - reset * VTH;
        mm[k] = nm;
        mx = fmaxf(mx, (nm > VTH) ? 1.f : 0.f);
    }
    *mv = make_float4(mm[0], mm[1], mm[2], mm[3]);
    p[(size_t)t * LP + pp] = mx;
}

// ---------------- per step: conv2 + bn2 + LIF2 -> s2 (f16 spikes) ------
__global__ void k_conv2_lif2(const float* __restrict__ p,
                             const float* __restrict__ w2, const float* __restrict__ cb,
                             const float* __restrict__ g, const float* __restrict__ bb,
                             const float* __restrict__ rm, const float* __restrict__ rv,
                             float* __restrict__ m2, _Float16* __restrict__ s2) {
    __shared__ float sw[C2 * C1 * 3];
    for (int i = threadIdx.x; i < C2 * C1 * 3; i += 256) sw[i] = w2[i];
    __syncthreads();
    int idx = blockIdx.x * 256 + threadIdx.x;
    if (idx >= NB * C2 * L2L) return;
    int l = idx % L2L; int t = idx / L2L;          // t = b*32 + oc
    int oc = t & 31;   int b = t >> 5;
    const float* pb = p + (size_t)b * C1 * LP + l;
    const float* wp = sw + oc * 48;
    float s = 0.f;
#pragma unroll
    for (int ic = 0; ic < C1; ++ic) {
        const float* pr = pb + ic * LP;
        s = fmaf(pr[0], wp[ic * 3 + 0], s);
        s = fmaf(pr[1], wp[ic * 3 + 1], s);
        s = fmaf(pr[2], wp[ic * 3 + 2], s);
    }
    float inv = g[oc] * rsqrtf(rv[oc] + BNEPS);
    float c2  = fmaf(s + cb[oc], inv, bb[oc] - rm[oc] * inv);
    int mi = idx;                                  // (b*32+oc)*493 + l == b*15776 + oc*493 + l
    float mem = m2[mi];
    float reset = (mem > VTH) ? 1.f : 0.f;
    mem = fmaf(BETA, mem, c2) - reset * VTH;
    m2[mi] = mem;
    s2[mi] = (_Float16)((mem > VTH) ? 1.f : 0.f);
}

// ---------------- per step: fc1 GEMM via WMMA f16, split-K=4 -----------
// S: spikes [128][15776] f16 (A, row-major). W: fc1_w [256][15776] f16 (B, N-major).
// Hp: partial sums [4][128][256] f32.
__global__ void k_fc1_wmma(const _Float16* __restrict__ S,
                           const _Float16* __restrict__ W,
                           float* __restrict__ Hp) {
    const int wave = threadIdx.x >> 5;
    const int lane = threadIdx.x & 31;
    const int mt = blockIdx.x & 7;       // 8 M-tiles of 16 rows
    const int kc = blockIdx.x >> 3;      // 4 K-chunks
    const int m0 = mt * 16;
    const int n0 = wave * 32;            // each wave: 16x32 output (two 16x16 tiles)
    const int lrow = lane & 15;
    const int lhi  = lane >> 4;
    const int kb = kc * 124;
    const int ke = (kb + 124 < KIT) ? kb + 124 : KIT;

    // Per ISA fragment tables (16-bit, 16x32 A / 32x16 B), lane (lrow, g=lhi)
    // holds half-groups K[8g..8g+8) and K[16+8g..16+8g+8) of its row/column.
    const _Float16* ap  = S + (size_t)(m0 + lrow) * NF + 8 * lhi;
    const _Float16* bp0 = W + (size_t)(n0 + lrow) * NF + 8 * lhi;
    const _Float16* bp1 = bp0 + (size_t)16 * NF;

    v8f acc0 = {}; v8f acc1 = {};
    union Frag { v16h v; v8h h[2]; };
    for (int ki = kb; ki < ke; ++ki) {
        const int ko = ki * 32;
        Frag a, b0, b1;
        a.h[0]  = *(const v8h*)(ap  + ko);
        a.h[1]  = *(const v8h*)(ap  + ko + 16);
        b0.h[0] = *(const v8h*)(bp0 + ko);
        b0.h[1] = *(const v8h*)(bp0 + ko + 16);
        b1.h[0] = *(const v8h*)(bp1 + ko);
        b1.h[1] = *(const v8h*)(bp1 + ko + 16);
        acc0 = __builtin_amdgcn_wmma_f32_16x16x32_f16(false, a.v, false, b0.v,
                                                      (short)0, acc0, false, false);
        acc1 = __builtin_amdgcn_wmma_f32_16x16x32_f16(false, a.v, false, b1.v,
                                                      (short)0, acc1, false, false);
    }
    float* out = Hp + (size_t)kc * NB * NH;
#pragma unroll
    for (int r = 0; r < 8; ++r) {
        int row = m0 + r + 8 * lhi;                 // C/D layout: VGPR r, lanes16-31 -> M=r+8
        out[row * NH + n0 + lrow]      = acc0[r];
        out[row * NH + n0 + 16 + lrow] = acc1[r];
    }
}

// ---------------- per step: sum split-K partials + LIF-H + pathway -----
__global__ void k_lifh(const float* __restrict__ Hp, const float* __restrict__ fb,
                       float* __restrict__ mh, float* __restrict__ path) {
    int i = blockIdx.x * 256 + threadIdx.x;
    if (i >= NB * NH) return;
    int j = i & (NH - 1);
    float h = Hp[i] + Hp[NB * NH + i] + Hp[2 * NB * NH + i] + Hp[3 * NB * NH + i] + fb[j];
    float mem = mh[i];
    float reset = (mem > VTH) ? 1.f : 0.f;
    mem = fmaf(BETA, mem, h) - reset * VTH;
    mh[i] = mem;
    path[i] += (mem > VTH) ? 1.f : 0.f;
}

// ---------------- final head: siamese linear + L2 normalize ------------
__global__ void k_head(const float* __restrict__ path, const float* __restrict__ sw,
                       const float* __restrict__ sb, float* __restrict__ out) {
    __shared__ float sp[NH];
    __shared__ float red[NE];
    int b = blockIdx.x; int j = threadIdx.x;       // 128 threads
    sp[j]        = path[b * NH + j];
    sp[j + NE]   = path[b * NH + j + NE];
    __syncthreads();
    float s = sb[j];
    const float* wr = sw + j * NH;
#pragma unroll 4
    for (int k = 0; k < NH; ++k) s = fmaf(sp[k], wr[k], s);
    red[j] = s * s;
    __syncthreads();
    for (int st = NE / 2; st > 0; st >>= 1) {
        if (j < st) red[j] += red[j + st];
        __syncthreads();
    }
    float norm = fmaxf(sqrtf(red[0]), 1e-12f);
    out[b * NE + j] = s / norm;
}

extern "C" void kernel_launch(void* const* d_in, const int* in_sizes, int n_in,
                              void* d_out, int out_size, void* d_ws, size_t ws_size,
                              hipStream_t stream) {
    const float* x[2]    = {(const float*)d_in[0], (const float*)d_in[1]};
    const float* conv1_w = (const float*)d_in[2];
    const float* conv1_b = (const float*)d_in[3];
    const float* bn1_g   = (const float*)d_in[4];
    const float* bn1_b   = (const float*)d_in[5];
    const float* bn1_m   = (const float*)d_in[6];
    const float* bn1_v   = (const float*)d_in[7];
    const float* conv2_w = (const float*)d_in[8];
    const float* conv2_b = (const float*)d_in[9];
    const float* bn2_g   = (const float*)d_in[10];
    const float* bn2_b   = (const float*)d_in[11];
    const float* bn2_m   = (const float*)d_in[12];
    const float* bn2_v   = (const float*)d_in[13];
    const float* fc1_w   = (const float*)d_in[14];
    const float* fc1_b   = (const float*)d_in[15];
    const float* sia_w   = (const float*)d_in[16];
    const float* sia_b   = (const float*)d_in[17];
    float* out = (float*)d_out;

    // Workspace carve (256B aligned)
    char* base = (char*)d_ws; size_t off = 0;
    auto carve = [&](size_t bytes) -> void* {
        void* p = base + off; off += (bytes + 255) & ~(size_t)255; return p;
    };
    _Float16* w16  = (_Float16*)carve((size_t)NF * NH * 2);        // fp16 weights
    float*    a1   = (float*)   carve((size_t)NB * C1 * L1P * 4);  // static drive
    float*    m1   = (float*)   carve((size_t)NB * C1 * L1P * 4);  // LIF1 state
    float*    pbuf = (float*)   carve((size_t)NB * C1 * LP * 4);   // pooled spikes
    float*    m2   = (float*)   carve((size_t)NB * C2 * L2L * 4);  // LIF2 state
    _Float16* s2   = (_Float16*)carve((size_t)NB * NF * 2);        // f16 spikes (A matrix)
    float*    Hp   = (float*)   carve((size_t)4 * NB * NH * 4);    // split-K partials
    float*    mh   = (float*)   carve((size_t)NB * NH * 4);        // LIF-H state
    float*    path = (float*)   carve((size_t)NB * NH * 4);        // pathway accumulator

    k_w2half<<<(NF * NH + 255) / 256, 256, 0, stream>>>(fc1_w, w16, NF * NH);

    for (int which = 0; which < 2; ++which) {
        hipMemsetAsync(m1,   0, (size_t)NB * C1 * L1P * 4, stream);
        hipMemsetAsync(m2,   0, (size_t)NB * C2 * L2L * 4, stream);
        hipMemsetAsync(mh,   0, (size_t)NB * NH * 4, stream);
        hipMemsetAsync(path, 0, (size_t)NB * NH * 4, stream);

        k_conv1<<<(NB * C1 * L1 + 255) / 256, 256, 0, stream>>>(
            x[which], conv1_w, conv1_b, bn1_g, bn1_b, bn1_m, bn1_v, a1);

        for (int s = 0; s < STEPS; ++s) {
            k_lif1_pool<<<(NB * C1 * LP + 255) / 256, 256, 0, stream>>>(a1, m1, pbuf);
            k_conv2_lif2<<<(NB * C2 * L2L + 255) / 256, 256, 0, stream>>>(
                pbuf, conv2_w, conv2_b, bn2_g, bn2_b, bn2_m, bn2_v, m2, s2);
            k_fc1_wmma<<<32, 256, 0, stream>>>(s2, w16, Hp);
            k_lifh<<<(NB * NH + 255) / 256, 256, 0, stream>>>(Hp, fc1_b, mh, path);
        }
        k_head<<<NB, NE, 0, stream>>>(path, sia_w, sia_b, out + (size_t)which * NB * NE);
    }
}
